// GptOssExperts_75557064671536
// MI455X (gfx1250) — matrix operand order repaired and checked by reading, not compiled, not run on hardware
//
#include <hip/hip_runtime.h>
#include <hip/hip_bf16.h>

// ---------------------------------------------------------------------------
// GPT-OSS MoE experts for MI455X (gfx1250).
//   GEMM1: hidden(8192x2880) x gate_up[e](2880x5760) -> interleaved gate/up
//   act  : bias + clamp + glu, stored f16 in workspace (8192x2880)
//   GEMM2: act x down[e](2880x2880) + bias -> out (f32)
// Compute: v_wmma_f32_16x16x32_f16 (f32 accum); f32->f16 convert on stage.
// Data movement: double-buffered LDS tiles, one LDS-only barrier per K-step
// (s_wait_dscnt + s_barrier_signal/-wait, so global loads stay in flight
// across barriers); GEMM2 A-tile moved by the Tensor Data Mover, pipelined
// one tile ahead (s_wait_tensorcnt 1), ragged tail zero-filled by TDM OOB.
// Memory floor ~77us @ 23.3 TB/s; f16 WMMA keeps compute near that floor.
// ---------------------------------------------------------------------------

typedef __attribute__((ext_vector_type(16))) _Float16 v16h;
typedef __attribute__((ext_vector_type(8)))  _Float16 v8h;
typedef __attribute__((ext_vector_type(4)))  _Float16 v4h;
typedef __attribute__((ext_vector_type(8)))  float    v8f;
typedef __attribute__((ext_vector_type(4)))  unsigned int u32x4;
typedef __attribute__((ext_vector_type(8)))  int      i32x8;
typedef __attribute__((ext_vector_type(4)))  int      i32x4;

#define T_TOK   8192
#define NEXP    16
#define HDIM    2880
#define DDIM    2880
#define GUPC    (2*DDIM)        // 5760
#define BM      128
#define BK      32
#define MAXT    (T_TOK/BM)      // 64 tiles worst case per expert

#if defined(__has_builtin)
#if __has_builtin(__builtin_amdgcn_tensor_load_to_lds)
#define HAVE_TDM 1
#endif
#endif

// Workgroup barrier that only drains LDS (dscnt): global loads issued before
// the barrier keep flying and are consumed by staggered s_wait_loadcnt at
// their converts. Mnemonics verified in prior rounds' disassembly.
__device__ __forceinline__ void barrier_lds() {
    asm volatile(
        "s_wait_dscnt 0x0\n"
        "s_barrier_signal -1\n"
        "s_barrier_wait -1"
        ::: "memory");
}

// ===========================================================================
// Kernel 1: ragged gate_up GEMM + biased clamped GLU activation -> act (f16)
// Block tile: 128 rows x 128 interleaved gate_up cols (= 64 act cols).
// 8 waves; wave (wm in 0..1, wn in 0..3) owns 4x2 16x16 WMMA subtiles.
// ===========================================================================
__global__ __launch_bounds__(256)
void gateup_act_kernel(const float* __restrict__ hidden,
                       const long long* __restrict__ etok,
                       const float* __restrict__ gup,
                       const float* __restrict__ gup_bias,
                       _Float16* __restrict__ act)
{
    const int e = blockIdx.x >> 6;
    const int t = blockIdx.x & 63;
    const long long start = (e == 0) ? 0LL : etok[e - 1];
    const long long end   = etok[e];
    const long long row_start = start + (long long)t * BM;
    if (row_start >= end) return;
    const long long rem = end - row_start;
    const int rows = (rem < (long long)BM) ? (int)rem : BM;

    __shared__ _Float16 sA[2][BM * BK];    // row-major: row*32 + k
    __shared__ _Float16 sB[2][128 * BK];   // col-major: col*32 + k

    const int tid  = threadIdx.x;
    const int lane = tid & 31;
    const int wave = tid >> 5;
    const int wm   = wave >> 2;         // 0..1 : 64-row half
    const int wn   = wave & 3;          // 0..3 : 32-col slice
    const int l15  = lane & 15;
    const int half = lane >> 4;

    const int cb = blockIdx.y * 128;    // base column in interleaved gate_up

    v8f acc[4][2];
    #pragma unroll
    for (int i = 0; i < 4; ++i)
        #pragma unroll
        for (int j = 0; j < 2; ++j)
            acc[i][j] = (v8f){0.f,0.f,0.f,0.f,0.f,0.f,0.f,0.f};

    const float* Abase = hidden + row_start * (long long)HDIM;
    const float* Bbase = gup + (long long)e * HDIM * GUPC + cb;

    // Per-thread staging geometry (fixed across iterations).
    int ar[4], ac[4], br[4], bc[4];
    float am[4];                        // branchless row mask
    #pragma unroll
    for (int c = 0; c < 4; ++c) {
        const int ia = tid + c * 256;
        ar[c] = ia >> 3;                // 0..127
        ac[c] = (ia & 7) * 4;           // 0..28
        am[c] = (ar[c] < rows) ? 1.f : 0.f;
        br[c] = ia >> 5;                // 0..31
        bc[c] = (ia & 31) * 4;          // 0..124
    }

    // Issue one tile's worth of global loads (no waits interleaved).
    auto load_tile = [&](int k, float4* aV, float4* bV) {
        #pragma unroll
        for (int c = 0; c < 4; ++c) {
            const int rEff = (am[c] != 0.f) ? ar[c] : 0;
            aV[c] = *(const float4*)(Abase + (long long)rEff * HDIM + k + ac[c]);
        }
        #pragma unroll
        for (int c = 0; c < 4; ++c)
            bV[c] = *(const float4*)(Bbase + (long long)(k + br[c]) * GUPC + bc[c]);
    };
    // Convert + publish a tile to LDS (branchless masking).
    auto store_tile = [&](float4* aV, float4* bV, int p) {
        #pragma unroll
        for (int c = 0; c < 4; ++c) {
            const float4 v = aV[c];
            const float m = am[c];
            v4h h;
            h.x = (_Float16)(v.x * m); h.y = (_Float16)(v.y * m);
            h.z = (_Float16)(v.z * m); h.w = (_Float16)(v.w * m);
            *(v4h*)&sA[p][ar[c] * BK + ac[c]] = h;
        }
        #pragma unroll
        for (int c = 0; c < 4; ++c) {
            const float4 v = bV[c];
            sB[p][(bc[c] + 0) * BK + br[c]] = (_Float16)v.x;
            sB[p][(bc[c] + 1) * BK + br[c]] = (_Float16)v.y;
            sB[p][(bc[c] + 2) * BK + br[c]] = (_Float16)v.z;
            sB[p][(bc[c] + 3) * BK + br[c]] = (_Float16)v.w;
        }
    };

    float4 aR[4], bR[4];
    load_tile(0, aR, bR);

    int p = 0;
    for (int k = 0; k < HDIM; k += BK, p ^= 1) {
        store_tile(aR, bR, p);
        barrier_lds();                         // publish buf p (LDS-only wait)

        const bool more = (k + BK) < HDIM;
        float4 aN[4], bN[4];
        if (more) {
            load_tile(k + BK, aN, bN);         // in flight across the WMMAs
            __builtin_prefetch(Bbase + (long long)(k + 2 * BK) * GUPC + tid * 16, 0, 0);
        }

        // ---- fragments (two ds_load_b128 each, per ISA 16-bit layouts) ----
        v16h afrag[4];
        #pragma unroll
        for (int mt = 0; mt < 4; ++mt) {
            const int row = wm * 64 + mt * 16 + l15;
            const v8h* q = (const v8h*)&sA[p][row * BK];
            const v8h lo = q[half];        // K0-7  (half=0) / K8-15  (half=1)
            const v8h hi = q[2 + half];    // K16-23(half=0) / K24-31 (half=1)
            afrag[mt] = __builtin_shufflevector(lo, hi,
                0,1,2,3,4,5,6,7,8,9,10,11,12,13,14,15);
        }
        v16h bfrag[2];
        #pragma unroll
        for (int nt = 0; nt < 2; ++nt) {
            const int col = wn * 32 + nt * 16 + l15;
            const v8h* q = (const v8h*)&sB[p][col * BK];
            const v8h lo = q[half];            // K0-7   / K8-15
            const v8h hi = q[2 + half];        // K16-23 / K24-31
            bfrag[nt] = __builtin_shufflevector(lo, hi,
                0,1,2,3,4,5,6,7,8,9,10,11,12,13,14,15);
        }

        #pragma unroll
        for (int mt = 0; mt < 4; ++mt)
            #pragma unroll
            for (int nt = 0; nt < 2; ++nt)
                acc[mt][nt] = __builtin_amdgcn_wmma_f32_16x16x32_f16(
                    false, afrag[mt], false, bfrag[nt],
                    (short)0, acc[mt][nt], false, false);

        if (more) {
            #pragma unroll
            for (int c = 0; c < 4; ++c) { aR[c] = aN[c]; bR[c] = bN[c]; }
        }
        // No trailing barrier: next iteration writes the other buffer; the
        // per-wave s_wait_dscnt at the next barrier_lds orders old readers.
    }

    // ---- epilogue: bias, gate/up pairing via shfl_xor(1), activation ----
    const float* biasp = gup_bias + (long long)e * GUPC + cb;
    #pragma unroll
    for (int mt = 0; mt < 4; ++mt) {
        #pragma unroll
        for (int nt = 0; nt < 2; ++nt) {
            const int ct = wn * 32 + nt * 16 + l15;   // tile-local gate_up col
            const float bv = biasp[ct];
            const v8f v = acc[mt][nt] + bv;
            #pragma unroll
            for (int r = 0; r < 8; ++r) {
                const float x    = v[r];
                const float part = __shfl_xor(x, 1, 32);
                float g = (lane & 1) ? part : x;       // even cols = gate
                float u = (lane & 1) ? x : part;       // odd cols  = up
                g = fminf(g, 7.f);
                u = fminf(fmaxf(u, -7.f), 7.f);
                const float glu = g / (1.f + __expf(-1.702f * g));
                const float a   = (u + 1.f) * glu;
                const int rloc  = wm * 64 + mt * 16 + r + half * 8;
                if (!(lane & 1) && rloc < rows) {
                    const long long rowg = row_start + rloc;
                    act[rowg * (long long)DDIM + (blockIdx.y * 64 + (ct >> 1))]
                        = (_Float16)a;
                }
            }
        }
    }
}

// ===========================================================================
// Kernel 2: ragged down-proj GEMM: act(f16) x down[e](f32->f16) + bias -> f32
// Block tile 128x64; 8 waves: wave (wm 0..1, wn 0..3) owns 4x1 WMMA subtiles.
// A-tile (pure f16 copy) moved by the Tensor Data Mover, double buffered and
// issued one tile ahead; ragged tail rows zero-filled by TDM OOB clipping.
// ===========================================================================
__global__ __launch_bounds__(256)
void down_kernel(const _Float16* __restrict__ act,
                 const long long* __restrict__ etok,
                 const float* __restrict__ down,
                 const float* __restrict__ down_bias,
                 float* __restrict__ out)
{
    const int e = blockIdx.x >> 6;
    const int t = blockIdx.x & 63;
    const long long start = (e == 0) ? 0LL : etok[e - 1];
    const long long end   = etok[e];
    const long long row_start = start + (long long)t * BM;
    if (row_start >= end) return;
    const long long rem = end - row_start;
    const int rows = (rem < (long long)BM) ? (int)rem : BM;

    __shared__ _Float16 sA[2][BM * BK];    // row-major (TDM destination)
    __shared__ _Float16 sB[2][64 * BK];    // col-major

    const int tid  = threadIdx.x;
    const int lane = tid & 31;
    const int wave = tid >> 5;
    const int wm   = wave >> 2;
    const int wn   = wave & 3;
    const int l15  = lane & 15;
    const int half = lane >> 4;

    const int cb = blockIdx.y * 64;

    v8f acc[4];
    #pragma unroll
    for (int i = 0; i < 4; ++i)
        acc[i] = (v8f){0.f,0.f,0.f,0.f,0.f,0.f,0.f,0.f};

    const _Float16* Abase = act + row_start * (long long)DDIM;
    const float*    Bbase = down + (long long)e * DDIM * HDIM + cb;

    // B staging geometry
    int br[2], bc[2];
    #pragma unroll
    for (int c = 0; c < 2; ++c) {
        const int idx = tid + c * 256;
        br[c] = idx >> 4;           // 0..31
        bc[c] = (idx & 15) * 4;     // 0..60
    }
    auto load_b = [&](int k, float4* bV) {
        #pragma unroll
        for (int c = 0; c < 2; ++c)
            bV[c] = *(const float4*)(Bbase + (long long)(k + br[c]) * HDIM + bc[c]);
    };
    auto store_b = [&](float4* bV, int p) {
        #pragma unroll
        for (int c = 0; c < 2; ++c) {
            const float4 v = bV[c];
            sB[p][(bc[c] + 0) * BK + br[c]] = (_Float16)v.x;
            sB[p][(bc[c] + 1) * BK + br[c]] = (_Float16)v.y;
            sB[p][(bc[c] + 2) * BK + br[c]] = (_Float16)v.z;
            sB[p][(bc[c] + 3) * BK + br[c]] = (_Float16)v.w;
        }
    };

#ifdef HAVE_TDM
    // D# group1 is loop-invariant: 2D tile 32(x) x 128(y), 2-byte elements,
    // row stride 2880 elements, y clipped at `rows` (OOB rows -> zeros).
    const i32x8 g1 = {
        (int)(1u << 16),                               // data_size=2B
        (int)(((unsigned)HDIM & 0xFFFFu) << 16),       // tensor_dim0 lo16
        (int)(((unsigned)rows) << 16),                 // dim0 hi16=0 | tensor_dim1 lo16
        (int)(32u << 16),                              // dim1 hi16=0 | tile_dim0=32
        128,                                           // tile_dim1=128, tile_dim2=0
        HDIM,                                          // tensor_dim0_stride lo32
        0, 0
    };
    const i32x4 gz4 = {0, 0, 0, 0};
#if __clang_major__ >= 23
    const i32x8 gz8 = {0, 0, 0, 0, 0, 0, 0, 0};
#endif
    unsigned lds_a[2];
    lds_a[0] = (unsigned)(uintptr_t)(&sA[0][0]);
    lds_a[1] = (unsigned)(uintptr_t)(&sA[1][0]);
    auto tdm_load_a = [&](int k, int p) {
        const unsigned long long ga =
            (unsigned long long)(uintptr_t)(const void*)(Abase + k);
        const u32x4 g0 = {
            1u,                                        // count=1
            lds_a[p],                                  // lds_addr
            (unsigned)(ga & 0xFFFFFFFFu),              // global_addr lo32
            (unsigned)(((ga >> 32) & 0x1FFFFFFu) | (2u << 30))  // hi25 | type=2
        };
#if __clang_major__ >= 23
        __builtin_amdgcn_tensor_load_to_lds(g0, g1, gz4, gz4, gz8, 0);
#else
        __builtin_amdgcn_tensor_load_to_lds(g0, g1, gz4, gz4, 0);
#endif
    };
#endif

    float4 bR[2];
    load_b(0, bR);
#ifdef HAVE_TDM
    if (wave == 0) tdm_load_a(0, 0);       // DMA tile 0 into buffer 0
#endif

    int p = 0;
    for (int k = 0; k < DDIM; k += BK, p ^= 1) {
        const bool more = (k + BK) < DDIM;

        store_b(bR, p);
#ifdef HAVE_TDM
        if (wave == 0) {
            if (more) tdm_load_a(k + BK, p ^ 1);   // DMA next tile ahead
            if (more) __builtin_amdgcn_s_wait_tensorcnt(1);  // tile k done
            else      __builtin_amdgcn_s_wait_tensorcnt(0);
        }
#else
        {
            const int r   = tid >> 1;
            const int off = (tid & 1) * 16;
            v8h v0 = {0,0,0,0,0,0,0,0};
            v8h v1 = {0,0,0,0,0,0,0,0};
            if (r < rows) {
                const v8h* gp = (const v8h*)(Abase + (long long)r * DDIM + k + off);
                v0 = gp[0];
                v1 = gp[1];
            }
            *(v8h*)&sA[p][r * BK + off]     = v0;
            *(v8h*)&sA[p][r * BK + off + 8] = v1;
        }
#endif
        barrier_lds();                          // publish buf p

        float4 bN[2];
        if (more) {
            load_b(k + BK, bN);
            __builtin_prefetch(Bbase + (long long)(k + 2 * BK) * HDIM + tid * 8, 0, 0);
        }

        v16h bfrag;
        {
            const int col = wn * 16 + l15;
            const v8h* q = (const v8h*)&sB[p][col * BK];
            const v8h lo = q[half];
            const v8h hi = q[2 + half];
            bfrag = __builtin_shufflevector(lo, hi,
                0,1,2,3,4,5,6,7,8,9,10,11,12,13,14,15);
        }
        #pragma unroll
        for (int mt = 0; mt < 4; ++mt) {
            const int row = wm * 64 + mt * 16 + l15;
            const v8h* q = (const v8h*)&sA[p][row * BK];
            const v8h lo = q[half];
            const v8h hi = q[2 + half];
            const v16h afrag = __builtin_shufflevector(lo, hi,
                0,1,2,3,4,5,6,7,8,9,10,11,12,13,14,15);
            acc[mt] = __builtin_amdgcn_wmma_f32_16x16x32_f16(
                false, afrag, false, bfrag, (short)0, acc[mt], false, false);
        }

        if (more) { bR[0] = bN[0]; bR[1] = bN[1]; }
    }

    // ---- epilogue: bias + f32 store ----
    const int col = cb + wn * 16 + l15;
    const float bv = down_bias[(long long)e * HDIM + col];
    #pragma unroll
    for (int mt = 0; mt < 4; ++mt) {
        #pragma unroll
        for (int r = 0; r < 8; ++r) {
            const int rloc = wm * 64 + mt * 16 + r + half * 8;
            if (rloc < rows) {
                const long long rowg = row_start + rloc;
                out[rowg * (long long)HDIM + col] = acc[mt][r] + bv;
            }
        }
    }
}

// ===========================================================================
extern "C" void kernel_launch(void* const* d_in, const int* in_sizes, int n_in,
                              void* d_out, int out_size, void* d_ws, size_t ws_size,
                              hipStream_t stream) {
    (void)in_sizes; (void)n_in; (void)out_size; (void)ws_size;
    const float*     hidden    = (const float*)d_in[0];
    const long long* etok      = (const long long*)d_in[1];
    const float*     gup       = (const float*)d_in[2];
    const float*     gup_bias  = (const float*)d_in[3];
    const float*     down      = (const float*)d_in[4];
    const float*     down_bias = (const float*)d_in[5];
    float*           out       = (float*)d_out;
    _Float16*        act       = (_Float16*)d_ws;   // 8192*2880 f16 = 47.2 MB

    const dim3 blk(256);
    const dim3 gA(NEXP * MAXT, DDIM / 64);   // 1024 x 45
    const dim3 gB(NEXP * MAXT, HDIM / 64);   // 1024 x 45

    gateup_act_kernel<<<gA, blk, 0, stream>>>(hidden, etok, gup, gup_bias, act);
    down_kernel<<<gB, blk, 0, stream>>>(act, etok, down, down_bias, out);
}
